// LSTMBase_3650722202007
// MI455X (gfx1250) — matrix operand order repaired and checked by reading, not compiled, not run on hardware
//
#include <hip/hip_runtime.h>
#include <hip/hip_bf16.h>

#define T_STEPS 512
#define BATCH   256
#define INDIM   512
#define HDIM    256
#define GDIM    1024   // 4*H

// padded LDS strides (elements) to avoid 64-bank conflicts
#define XS 520   // x tile, bf16
#define HS 264   // h tile, bf16
#define CS 260   // c / h f32
#define GS 1028  // gates f32

typedef __bf16        v16bf __attribute__((ext_vector_type(16)));
typedef float         v8f   __attribute__((ext_vector_type(8)));
typedef unsigned int  v4u   __attribute__((ext_vector_type(4)));  // direct VReg_128

union BfOp { v16bf v; v4u q[2]; };

__device__ __forceinline__ unsigned short f2bf(float f) {
    unsigned int u = __float_as_uint(f);
    u += 0x7FFFu + ((u >> 16) & 1u);          // round-to-nearest-even
    return (unsigned short)(u >> 16);
}
__device__ __forceinline__ float sigm(float x) { return 1.0f / (1.0f + __expf(-x)); }
__device__ __forceinline__ float tanh_fast(float x) {
    x = fminf(fmaxf(x, -15.0f), 15.0f);
    float e = __expf(-2.0f * x);
    return (1.0f - e) / (1.0f + e);
}

// one-shot: W -> bf16 into workspace, bias = b_ih + b_hh
__global__ void lstm_prep_kernel(const float* __restrict__ Wih, const float* __restrict__ Whh,
                                 const float* __restrict__ bih, const float* __restrict__ bhh,
                                 unsigned short* __restrict__ wih_bf,
                                 unsigned short* __restrict__ whh_bf,
                                 float* __restrict__ bias) {
    int i = blockIdx.x * blockDim.x + threadIdx.x;
    const int NW1 = GDIM * INDIM;
    const int NW2 = GDIM * HDIM;
    if (i < NW1) {
        wih_bf[i] = f2bf(Wih[i]);
    } else if (i < NW1 + NW2) {
        whh_bf[i - NW1] = f2bf(Whh[i - NW1]);
    } else if (i < NW1 + NW2 + GDIM) {
        int j = i - NW1 - NW2;
        bias[j] = bih[j] + bhh[j];
    }
}

// ---- asm B-fragment loads: volatile => never hoisted/spilled by LICM ----
// 32 contiguous bytes of one weight row per lane; compile-time offsets.
#define GLOAD2(d0, d1, base, off)                                             \
    asm volatile("global_load_b128 %0, %2, off offset:%3\n\t"                 \
                 "global_load_b128 %1, %2, off offset:%4"                     \
                 : "=&v"(d0), "=&v"(d1)                                       \
                 : "v"(base), "i"(off), "i"((off) + 16))

#define LOADB8(Bb, base, ROWB, kbyte)                                         \
    do {                                                                      \
        GLOAD2(Bb[0].q[0], Bb[0].q[1], base, (kbyte));                        \
        GLOAD2(Bb[1].q[0], Bb[1].q[1], base, (kbyte) + 1 * (ROWB));           \
        GLOAD2(Bb[2].q[0], Bb[2].q[1], base, (kbyte) + 2 * (ROWB));           \
        GLOAD2(Bb[3].q[0], Bb[3].q[1], base, (kbyte) + 3 * (ROWB));           \
        GLOAD2(Bb[4].q[0], Bb[4].q[1], base, (kbyte) + 4 * (ROWB));           \
        GLOAD2(Bb[5].q[0], Bb[5].q[1], base, (kbyte) + 5 * (ROWB));           \
        GLOAD2(Bb[6].q[0], Bb[6].q[1], base, (kbyte) + 6 * (ROWB));           \
        GLOAD2(Bb[7].q[0], Bb[7].q[1], base, (kbyte) + 7 * (ROWB));           \
    } while (0)

// Wait until <= N asm loads outstanding; "+v" ties (direct ext-vector regs)
// make the consuming WMMAs data-depend on the wait.
#define WAITB(Bb, N)                                                          \
    asm volatile("s_wait_loadcnt " #N                                         \
                 : "+v"(Bb[0].q[0]), "+v"(Bb[0].q[1]),                        \
                   "+v"(Bb[1].q[0]), "+v"(Bb[1].q[1]),                        \
                   "+v"(Bb[2].q[0]), "+v"(Bb[2].q[1]),                        \
                   "+v"(Bb[3].q[0]), "+v"(Bb[3].q[1]),                        \
                   "+v"(Bb[4].q[0]), "+v"(Bb[4].q[1]),                        \
                   "+v"(Bb[5].q[0]), "+v"(Bb[5].q[1]),                        \
                   "+v"(Bb[6].q[0]), "+v"(Bb[6].q[1]),                        \
                   "+v"(Bb[7].q[0]), "+v"(Bb[7].q[1]))

// A fragment from LDS (compiler-managed dscnt waits)
#define LOAD_A(Adst, sbuf, stride, kb)                                        \
    do {                                                                      \
        Adst.q[0] = *(const v4u*)&sbuf[nlo * (stride) + (kb) + c0a];          \
        Adst.q[1] = *(const v4u*)&sbuf[nlo * (stride) + (kb) + c0a + 8];      \
    } while (0)

#define WMMA8(Abuf, Bbuf)                                                     \
    do {                                                                      \
        _Pragma("unroll")                                                     \
        for (int nt = 0; nt < 8; ++nt)                                        \
            acc[nt] = __builtin_amdgcn_wmma_f32_16x16x32_bf16(                \
                false, Abuf.v, false, Bbuf[nt].v, (short)0, acc[nt],          \
                false, false);                                                \
    } while (0)

// Persistent LSTM: each workgroup owns a 16-row batch tile for all T steps.
// 8 waves; wave w computes gate columns [128w, 128w+128) with bf16 WMMA.
__global__ __launch_bounds__(256) void lstm_persistent_kernel(
    const float* __restrict__ x_seq, const int* __restrict__ masks,
    const unsigned short* __restrict__ wih, const unsigned short* __restrict__ whh,
    const float* __restrict__ bias_g, const float* __restrict__ gamma_g,
    const float* __restrict__ beta_g, float* __restrict__ out) {

    __shared__ unsigned short s_x[16 * XS];    // x_t tile, bf16
    __shared__ unsigned short s_hA[16 * HS];   // h (masked), bf16 A operand
    __shared__ float s_hF[16 * CS];            // h, f32
    __shared__ float s_c [16 * CS];            // c, f32
    __shared__ float s_g [16 * GS];            // gates, f32
    __shared__ float s_red[2][16][16];
    __shared__ float s_ms[16][2];
    __shared__ float s_bias[GDIM];
    __shared__ float s_gam[HDIM];
    __shared__ float s_bet[HDIM];

    const int tid   = threadIdx.x;
    const int bbase = blockIdx.x * 16;
    const int row   = tid >> 4;     // 0..15 batch row within tile
    const int jj    = tid & 15;     // 16 threads per row, 16 cols each

    for (int i = 0; i < 16; ++i) {
        int col = jj * 16 + i;
        s_hF[row * CS + col] = 0.0f;
        s_c [row * CS + col] = 0.0f;
    }
    s_bias[tid]       = bias_g[tid];
    s_bias[tid + 256] = bias_g[tid + 256];
    s_bias[tid + 512] = bias_g[tid + 512];
    s_bias[tid + 768] = bias_g[tid + 768];
    s_gam[tid] = gamma_g[tid];
    s_bet[tid] = beta_g[tid];
    __syncthreads();

    const int wave  = tid >> 5;
    const int lane  = tid & 31;
    const int nlo   = lane & 15;           // A row m / B col n within tile
    const int c0a   = (lane >> 4) * 16;    // A K-chunk base (contiguous 16)
    const int c0b   = (lane >> 4) * 16;    // B K-chunk base (contiguous 16)
    const int nbase = wave * 128;

    // per-lane weight base addresses (row nbase+nlo, K-chunk c0b)
    const unsigned short* wb_ih = wih + (size_t)(nbase + nlo) * INDIM + c0b;
    const unsigned short* wb_hh = whh + (size_t)(nbase + nlo) * HDIM + c0b;

    for (int t = 0; t < T_STEPS; ++t) {
        // ---- mask-reset h,c ; build bf16 h operand ----
        float keep = 1.0f - (float)masks[t * BATCH + bbase + row];
        for (int i = 0; i < 16; ++i) {
            int col = jj * 16 + i;
            float hv = s_hF[row * CS + col] * keep;
            s_hF[row * CS + col] = hv;
            s_c [row * CS + col] *= keep;
            s_hA[row * HS + col] = f2bf(hv);
        }
        // ---- load x_t tile (coalesced float4), convert to bf16 ----
        const float* xr = x_seq + ((size_t)t * BATCH + bbase + row) * INDIM;
        for (int ch = 0; ch < 8; ++ch) {
            int col = jj * 4 + ch * 64;
            float4 v = *(const float4*)(xr + col);
            unsigned int p0 = ((unsigned int)f2bf(v.y) << 16) | f2bf(v.x);
            unsigned int p1 = ((unsigned int)f2bf(v.w) << 16) | f2bf(v.z);
            *(unsigned int*)&s_x[row * XS + col]     = p0;
            *(unsigned int*)&s_x[row * XS + col + 2] = p1;
        }
        if (t + 1 < T_STEPS)   // warm L2/WGP$ for next step's x tile
            __builtin_prefetch(xr + (size_t)BATCH * INDIM, 0, 1);
        __syncthreads();

        // ---- gates[16 x 1024] slice via v_wmma_f32_16x16x32_bf16 ----
        // Hand-pipelined: K-tile k+1's 16 B loads are in flight during
        // K-tile k's 8 WMMAs; one s_wait_loadcnt 0x10 per tile instead of
        // s_wait_loadcnt 0 per WMMA. asm-volatile loads defeat LICM/spills.
        v8f acc[8] = {};
        {
            BfOp A;
            BfOp B0[8], B1[8];
            // x @ W_ih^T : K = 512, 16 K-tiles, rowBytes*16 = 16384
            LOADB8(B0, wb_ih, 16384, 0);
#pragma unroll
            for (int kt = 0; kt < 16; kt += 2) {
                LOADB8(B1, wb_ih, 16384, (kt + 1) * 64);
                LOAD_A(A, s_x, XS, kt * 32);
                WAITB(B0, 0x10);
                WMMA8(A, B0);
                if (kt + 2 < 16) {
                    LOADB8(B0, wb_ih, 16384, (kt + 2) * 64);
                    LOAD_A(A, s_x, XS, (kt + 1) * 32);
                    WAITB(B1, 0x10);
                    WMMA8(A, B1);
                } else {
                    LOAD_A(A, s_x, XS, (kt + 1) * 32);
                    WAITB(B1, 0x0);
                    WMMA8(A, B1);
                }
            }
            // h @ W_hh^T : K = 256, 8 K-tiles, rowBytes*16 = 8192
            LOADB8(B0, wb_hh, 8192, 0);
#pragma unroll
            for (int kt = 0; kt < 8; kt += 2) {
                LOADB8(B1, wb_hh, 8192, (kt + 1) * 64);
                LOAD_A(A, s_hA, HS, kt * 32);
                WAITB(B0, 0x10);
                WMMA8(A, B0);
                if (kt + 2 < 8) {
                    LOADB8(B0, wb_hh, 8192, (kt + 2) * 64);
                    LOAD_A(A, s_hA, HS, (kt + 1) * 32);
                    WAITB(B1, 0x10);
                    WMMA8(A, B1);
                } else {
                    LOAD_A(A, s_hA, HS, (kt + 1) * 32);
                    WAITB(B1, 0x0);
                    WMMA8(A, B1);
                }
            }
        }
        // C/D layout: acc[nt][i] -> row (i + 8*(lane>=16)), col (lane&15)
        {
            int rb = (lane >> 4) * 8;
#pragma unroll
            for (int nt = 0; nt < 8; ++nt)
#pragma unroll
                for (int i = 0; i < 8; ++i)
                    s_g[(rb + i) * GS + nbase + nt * 16 + nlo] = acc[nt][i];
        }
        __syncthreads();

        // ---- cell update + LayerNorm stats ----
        float sum = 0.0f, sq = 0.0f;
        for (int i = 0; i < 16; ++i) {
            int col = jj * 16 + i;
            float gi = s_g[row * GS + col]       + s_bias[col];
            float gf = s_g[row * GS + 256 + col] + s_bias[256 + col];
            float gg = s_g[row * GS + 512 + col] + s_bias[512 + col];
            float go = s_g[row * GS + 768 + col] + s_bias[768 + col];
            float cc = sigm(gf) * s_c[row * CS + col] + sigm(gi) * tanh_fast(gg);
            s_c[row * CS + col] = cc;
            float hh = sigm(go) * tanh_fast(cc);
            s_hF[row * CS + col] = hh;
            sum += hh; sq += hh * hh;
        }
        s_red[0][row][jj] = sum;
        s_red[1][row][jj] = sq;
        __syncthreads();
        if (tid < 16) {
            float s = 0.0f, s2 = 0.0f;
            for (int k = 0; k < 16; ++k) { s += s_red[0][tid][k]; s2 += s_red[1][tid][k]; }
            float mu  = s  * (1.0f / HDIM);
            float var = s2 * (1.0f / HDIM) - mu * mu;
            s_ms[tid][0] = mu;
            s_ms[tid][1] = __frsqrt_rn(var + 1e-5f);
        }
        __syncthreads();
        float mu = s_ms[row][0], rs = s_ms[row][1];
        float* yr = out + ((size_t)t * BATCH + bbase + row) * HDIM;
        for (int i = 0; i < 16; ++i) {
            int col = jj * 16 + i;
            yr[col] = (s_hF[row * CS + col] - mu) * rs * s_gam[col] + s_bet[col];
        }
        __syncthreads();   // protect s_hF/s_c/s_x before next step's writes
    }

    // ---- final h, c (post-step-511, no further masking) ----
    float* hout = out + (size_t)T_STEPS * BATCH * HDIM;
    float* cout_ = hout + (size_t)BATCH * HDIM;
    for (int i = 0; i < 16; ++i) {
        int col = jj * 16 + i;
        hout [(size_t)(bbase + row) * HDIM + col] = s_hF[row * CS + col];
        cout_[(size_t)(bbase + row) * HDIM + col] = s_c [row * CS + col];
    }
}

extern "C" void kernel_launch(void* const* d_in, const int* in_sizes, int n_in,
                              void* d_out, int out_size, void* d_ws, size_t ws_size,
                              hipStream_t stream) {
    (void)in_sizes; (void)n_in; (void)out_size; (void)ws_size;
    const float* x_seq = (const float*)d_in[0];
    const int*   masks = (const int*)d_in[1];
    const float* W_ih  = (const float*)d_in[2];
    const float* W_hh  = (const float*)d_in[3];
    const float* b_ih  = (const float*)d_in[4];
    const float* b_hh  = (const float*)d_in[5];
    const float* gamma = (const float*)d_in[6];
    const float* beta  = (const float*)d_in[7];
    float* out = (float*)d_out;

    // workspace layout: [W_ih bf16 | W_hh bf16 | bias f32]  (~1.6 MB)
    unsigned short* wih_bf = (unsigned short*)d_ws;
    unsigned short* whh_bf = wih_bf + (size_t)GDIM * INDIM;
    float* bias = (float*)(whh_bf + (size_t)GDIM * HDIM);

    int total = GDIM * INDIM + GDIM * HDIM + GDIM;
    lstm_prep_kernel<<<(total + 255) / 256, 256, 0, stream>>>(
        W_ih, W_hh, b_ih, b_hh, wih_bf, whh_bf, bias);
    lstm_persistent_kernel<<<BATCH / 16, 256, 0, stream>>>(
        x_seq, masks, wih_bf, whh_bf, bias, gamma, beta, out);
}
